// STU_10977936408688
// MI455X (gfx1250) — compile-verified, matrix operand order checked by reading, not run on hardware
//
#include <hip/hip_runtime.h>
#include <hip/hip_bf16.h>

#define BSZ 4
#define SL 2048
#define DIM 256
#define NK 24
#define KUL 3
#define NFC 6                 // filters per chunk
#define NCHUNK 4              // NK / NFC
#define KC (NFC * DIM)        // 1536: per-chunk GEMM K
#define AST 40                // LDS tile row stride in u16 (32 data + 8 pad)
#define VPAD 128              // zero prefix for causal Toeplitz gather

typedef __attribute__((ext_vector_type(8)))  unsigned short us8;
typedef __attribute__((ext_vector_type(16))) unsigned short us16;
typedef __attribute__((ext_vector_type(16))) __bf16        v16bf;
typedef __attribute__((ext_vector_type(8)))  float         v8f;

__device__ __forceinline__ unsigned short f2bf(float x) {
  unsigned u = __builtin_bit_cast(unsigned, x);
  u += 0x7fffu + ((u >> 16) & 1u);          // round-to-nearest-even
  return (unsigned short)(u >> 16);
}

// Per-lane fragment gather per CDNA5 16-bit 16x32 layout:
// lanes 0-15 hold rows with K runs [0..7],[16..23]; lanes 16-31: [8..15],[24..31].
__device__ __forceinline__ v16bf load_frag(const unsigned short* row, int lh) {
  us8 x0 = *(const us8*)(row + 8 * lh);
  us8 x1 = *(const us8*)(row + 16 + 8 * lh);
  union { us16 u; v16bf v; } w;
#pragma unroll
  for (int j = 0; j < 8; ++j) { w.u[j] = x0[j]; w.u[j + 8] = x1[j]; }
  return w.v;
}

__device__ __forceinline__ v8f wmma_bf16(v16bf a, v16bf b, v8f c) {
  return __builtin_amdgcn_wmma_f32_16x16x32_bf16(false, a, false, b, (short)0, c,
                                                 false, false);
}

// ---------------------------------------------------------------------------
// Stage A: causal spectral convolution via Toeplitz WMMA.
// 128t x 64d macro-tile, 8 waves, 2x2 register blocking (4 wmma / wave / k).
// grid: (SL/128, DIM/64, BSZ*NFC), block 256. Writes xt[b][t][fl*DIM+d] bf16.
// ---------------------------------------------------------------------------
__global__ __launch_bounds__(256) void conv_kernel(
    const float* __restrict__ u, const float* __restrict__ evec,
    const float* __restrict__ eval, unsigned short* __restrict__ xt, int chunk) {
  __shared__ __align__(16) unsigned short vflp[VPAD + SL];  // zero-padded filter
  __shared__ __align__(16) unsigned short At[128 * AST];
  __shared__ __align__(16) unsigned short Bt[64 * AST];

  const int t0 = blockIdx.x * 128;
  const int d0 = blockIdx.y * 64;
  const int b  = blockIdx.z / NFC;
  const int fl = blockIdx.z % NFC;
  const int f  = chunk * NFC + fl;
  const int tid = threadIdx.x;

  // zero prefix + scaled filter -> LDS (unconditional gathers later)
  const float sc = __powf(eval[f], 0.25f);
  if (tid < VPAD) vflp[tid] = 0;
  for (int l = tid; l < SL; l += 256) vflp[VPAD + l] = f2bf(evec[l * NK + f] * sc);

  const int wave = tid >> 5, lane = tid & 31, lr = lane & 15, lh = lane >> 4;
  const int mp = (wave & 3) * 2;        // M-subtile pair
  const int np = (wave >> 2) * 2;       // N-subtile pair
  v8f a00 = {0.f, 0.f, 0.f, 0.f, 0.f, 0.f, 0.f, 0.f};
  v8f a01 = a00, a10 = a00, a11 = a00;

  const int fi = tid >> 2, fkb = (tid & 3) * 8;   // A-fill coords
  const int dn = tid & 63, bkb = (tid >> 6) * 8;  // B-fill coords
  const float* up = u + (size_t)b * SL * DIM + d0 + dn;

  const int nsc = (t0 >> 5) + 4;        // s-chunks covering s <= t0+127
  for (int c = 0; c < nsc; ++c) {
    const int s0 = c * 32;
    __syncthreads();
    // Toeplitz A tile: At[i][k] = vflp[VPAD + (t0+i) - (s0+k)] (zeros = causal)
#pragma unroll
    for (int r = 0; r < 2; ++r) {
      const int ii = fi + 64 * r;
      const int base = VPAD + t0 + ii - s0 - fkb;
#pragma unroll
      for (int j = 0; j < 8; ++j) At[ii * AST + fkb + j] = vflp[base - j];
    }
    // B tile (n-major): Bt[dn][k] = bf16(u[b][s0+k][d0+dn]); coalesced over dn
#pragma unroll
    for (int j = 0; j < 8; ++j)
      Bt[dn * AST + bkb + j] = f2bf(up[(size_t)(s0 + bkb + j) * DIM]);
    // prefetch next chunk's input rows into cache (clamped, in-buffer)
    {
      const int sp = s0 + 32 + bkb;
      const int spc = sp < SL ? sp : SL - 1;
      __builtin_prefetch(up + (size_t)spc * DIM, 0, 1);
    }
    __syncthreads();
    v16bf fA0 = load_frag(&At[(mp * 16 + lr) * AST], lh);
    v16bf fA1 = load_frag(&At[((mp + 1) * 16 + lr) * AST], lh);
    v16bf fB0 = load_frag(&Bt[(np * 16 + lr) * AST], lh);
    v16bf fB1 = load_frag(&Bt[((np + 1) * 16 + lr) * AST], lh);
    a00 = wmma_bf16(fA0, fB0, a00);
    a01 = wmma_bf16(fA0, fB1, a01);
    a10 = wmma_bf16(fA1, fB0, a10);
    a11 = wmma_bf16(fA1, fB1, a11);
  }

  const int cb = fl * DIM + d0 + np * 16 + lr;
#pragma unroll
  for (int r = 0; r < 8; ++r) {
    const int m0v = mp * 16 + r + 8 * lh;
    const size_t r0 = (size_t)(b * SL + t0 + m0v) * KC;
    const size_t r1 = (size_t)(b * SL + t0 + m0v + 16) * KC;
    xt[r0 + cb]      = f2bf(a00[r]);
    xt[r0 + cb + 16] = f2bf(a01[r]);
    xt[r1 + cb]      = f2bf(a10[r]);
    xt[r1 + cb + 16] = f2bf(a11[r]);
  }
}

// ---------------------------------------------------------------------------
// Stage B: deltas += xt_chunk @ m_phi_chunk.  grid: (BSZ*SL/128, DIM/64, 1).
// ---------------------------------------------------------------------------
__global__ __launch_bounds__(256) void gemm_phi_kernel(
    const unsigned short* __restrict__ xt, const float* __restrict__ mphi,
    float* __restrict__ deltas, int chunk) {
  __shared__ __align__(16) unsigned short At[128 * AST];
  __shared__ __align__(16) unsigned short Bt[64 * AST];

  const int m0 = blockIdx.x * 128;
  const int n0 = blockIdx.y * 64;
  const int tid = threadIdx.x;
  const int wave = tid >> 5, lane = tid & 31, lr = lane & 15, lh = lane >> 4;
  const int mp = (wave & 3) * 2, np = (wave >> 2) * 2;
  v8f a00 = {0.f, 0.f, 0.f, 0.f, 0.f, 0.f, 0.f, 0.f};
  v8f a01 = a00, a10 = a00, a11 = a00;

  const int fi = tid >> 1, fkb = (tid & 1) * 16;  // A: 128 rows, 32B per thread

  for (int kc = 0; kc < KC / 32; ++kc) {
    const int k0 = kc * 32;
    __syncthreads();
    {  // A: vectorized bf16 copy global->LDS (16B aligned)
      const unsigned short* src = xt + (size_t)(m0 + fi) * KC + k0 + fkb;
      *(us8*)&At[fi * AST + fkb]     = *(const us8*)src;
      *(us8*)&At[fi * AST + fkb + 8] = *(const us8*)(src + 8);
    }
    {  // B (n-major): Bt[n][k] = bf16(mphi[(chunk*KC+k0+k)*DIM + n0+n])
#pragma unroll
      for (int j = 0; j < 8; ++j) {
        const int e = tid + 256 * j;
        const int k = e >> 6, n = e & 63;
        Bt[n * AST + k] =
            f2bf(mphi[(size_t)(chunk * KC + k0 + k) * DIM + n0 + n]);
      }
    }
    // prefetch next k-chunk sources into cache
    if (kc + 1 < KC / 32) {
      __builtin_prefetch(xt + (size_t)(m0 + fi) * KC + k0 + 32 + fkb, 0, 1);
      __builtin_prefetch(
          mphi + (size_t)(chunk * KC + k0 + 32 + (tid >> 3)) * DIM + n0, 0, 1);
    }
    __syncthreads();
    v16bf fA0 = load_frag(&At[(mp * 16 + lr) * AST], lh);
    v16bf fA1 = load_frag(&At[((mp + 1) * 16 + lr) * AST], lh);
    v16bf fB0 = load_frag(&Bt[(np * 16 + lr) * AST], lh);
    v16bf fB1 = load_frag(&Bt[((np + 1) * 16 + lr) * AST], lh);
    a00 = wmma_bf16(fA0, fB0, a00);
    a01 = wmma_bf16(fA0, fB1, a01);
    a10 = wmma_bf16(fA1, fB0, a10);
    a11 = wmma_bf16(fA1, fB1, a11);
  }

  const int c0 = n0 + np * 16 + lr;
#pragma unroll
  for (int r = 0; r < 8; ++r) {
    const int mv = mp * 16 + r + 8 * lh;
    const size_t r0 = (size_t)(m0 + mv) * DIM;
    const size_t r1 = (size_t)(m0 + mv + 16) * DIM;
    deltas[r0 + c0]      += a00[r];
    deltas[r0 + c0 + 16] += a01[r];
    deltas[r1 + c0]      += a10[r];
    deltas[r1 + c0 + 16] += a11[r];
  }
}

// ---------------------------------------------------------------------------
// Stage C: AR term (initializes deltas). Implicit im2col over (d, lag), K=768.
// grid: (SL/128, DIM/64, BSZ). Clamped addresses, value select (no exec dance).
// ---------------------------------------------------------------------------
__global__ __launch_bounds__(256) void ar_kernel(
    const float* __restrict__ u, const float* __restrict__ mu,
    float* __restrict__ deltas) {
  __shared__ __align__(16) unsigned short At[128 * AST];
  __shared__ __align__(16) unsigned short Bt[64 * AST];

  const int t0 = blockIdx.x * 128;
  const int n0 = blockIdx.y * 64;
  const int b  = blockIdx.z;
  const int tid = threadIdx.x;
  const int wave = tid >> 5, lane = tid & 31, lr = lane & 15, lh = lane >> 4;
  const int mp = (wave & 3) * 2, np = (wave >> 2) * 2;
  v8f a00 = {0.f, 0.f, 0.f, 0.f, 0.f, 0.f, 0.f, 0.f};
  v8f a01 = a00, a10 = a00, a11 = a00;

  const int fi = tid >> 2, fkb = (tid & 3) * 8;

  for (int kc = 0; kc < (DIM * KUL) / 32; ++kc) {
    const int k0 = kc * 32;
    __syncthreads();
    // A: At[i][k] = bf16(u[b][t0+i-lag][d]), kflat = d*3+lag
#pragma unroll
    for (int r = 0; r < 2; ++r) {
      const int ii = fi + 64 * r;
#pragma unroll
      for (int j = 0; j < 8; ++j) {
        const int kg = k0 + fkb + j;
        const int d = kg / KUL, lag = kg - d * KUL;
        const int t = t0 + ii - lag;
        const int tc = t < 0 ? 0 : t;   // clamp address, select value
        const float v = u[((size_t)b * SL + tc) * DIM + d];
        At[ii * AST + fkb + j] = (t >= 0) ? f2bf(v) : (unsigned short)0;
      }
    }
    // B (n-major): Bt[n][k] = bf16(mu[(n0+n)*768 + k0+k])
#pragma unroll
    for (int j = 0; j < 8; ++j) {
      const int e = tid + 256 * j;
      const int k = e >> 6, n = e & 63;
      Bt[n * AST + k] = f2bf(mu[(size_t)(n0 + n) * (DIM * KUL) + k0 + k]);
    }
    // prefetch next k-chunk of the small weight matrix
    if (kc + 1 < (DIM * KUL) / 32)
      __builtin_prefetch(mu + (size_t)(n0 + (tid & 63)) * (DIM * KUL) + k0 + 32,
                         0, 1);
    __syncthreads();
    v16bf fA0 = load_frag(&At[(mp * 16 + lr) * AST], lh);
    v16bf fA1 = load_frag(&At[((mp + 1) * 16 + lr) * AST], lh);
    v16bf fB0 = load_frag(&Bt[(np * 16 + lr) * AST], lh);
    v16bf fB1 = load_frag(&Bt[((np + 1) * 16 + lr) * AST], lh);
    a00 = wmma_bf16(fA0, fB0, a00);
    a01 = wmma_bf16(fA0, fB1, a01);
    a10 = wmma_bf16(fA1, fB0, a10);
    a11 = wmma_bf16(fA1, fB1, a11);
  }

  const int c0 = n0 + np * 16 + lr;
#pragma unroll
  for (int r = 0; r < 8; ++r) {
    const int mv = mp * 16 + r + 8 * lh;
    const size_t r0 = (size_t)(b * SL + t0 + mv) * DIM;
    const size_t r1 = (size_t)(b * SL + t0 + mv + 16) * DIM;
    deltas[r0 + c0]      = a00[r];
    deltas[r0 + c0 + 16] = a01[r];
    deltas[r1 + c0]      = a10[r];
    deltas[r1 + c0 + 16] = a11[r];
  }
}

// ---------------------------------------------------------------------------
// Stage D: sequential AR(2) scan, one workgroup per batch; m_y cached in LDS
// as bf16 (256 KB <= 320 KB/WGP); per-lane rotation -> conflict-free banks.
// ---------------------------------------------------------------------------
__global__ __launch_bounds__(256) void scan_kernel(
    const float* __restrict__ deltas, const float* __restrict__ m_y,
    float* __restrict__ out) {
  __shared__ __align__(16) unsigned short Wl[DIM * 2 * DIM];  // 256 KB
  __shared__ __align__(16) float sv[2 * DIM];                 // [y_{t-1}|y_{t-2}]
  const int b = blockIdx.x;
  const int o = threadIdx.x;

  for (int e = o; e < DIM * 2 * DIM; e += 256) Wl[e] = f2bf(m_y[e]);
  sv[o] = 0.f;
  sv[o + DIM] = 0.f;
  __syncthreads();

  const float* dl = deltas + (size_t)b * SL * DIM;
  float* yo = out + (size_t)b * SL * DIM;
  const unsigned short* wr = Wl + o * 512;
  const int rot = (o * 2) & 511;  // even rotation: distinct banks across lanes

  for (int t = 0; t < SL; ++t) {
    float acc = dl[(size_t)t * DIM + o];
#pragma unroll 8
    for (int dd = 0; dd < 512; dd += 2) {
      const int d = (dd + rot) & 511;
      const unsigned wp = *(const unsigned*)(wr + d);
      const float w0 = __builtin_bit_cast(float, wp << 16);
      const float w1 = __builtin_bit_cast(float, wp & 0xffff0000u);
      const float2 s2 = *(const float2*)(sv + d);
      acc += w0 * s2.x + w1 * s2.y;
    }
    const float prev = sv[o];
    __syncthreads();
    sv[o] = acc;          // y_{t-1} <- y_t
    sv[o + DIM] = prev;   // y_{t-2} <- old y_{t-1}
    yo[(size_t)t * DIM + o] = acc;
    __syncthreads();
  }
}

// ---------------------------------------------------------------------------
extern "C" void kernel_launch(void* const* d_in, const int* in_sizes, int n_in,
                              void* d_out, int out_size, void* d_ws,
                              size_t ws_size, hipStream_t stream) {
  (void)in_sizes; (void)n_in; (void)out_size; (void)ws_size;
  const float* u    = (const float*)d_in[0];  // [4,2048,256]
  const float* mu   = (const float*)d_in[1];  // [256,256,3]
  const float* mphi = (const float*)d_in[2];  // [6144,256]
  const float* my   = (const float*)d_in[3];  // [256,2,256]
  const float* eval = (const float*)d_in[4];  // [24]
  const float* evec = (const float*)d_in[5];  // [2048,24]
  float* out = (float*)d_out;

  // workspace: deltas f32 (8 MB) | xt bf16 chunk (24 MB)  => 32 MB total
  float* deltas = (float*)d_ws;
  unsigned short* xt =
      (unsigned short*)((char*)d_ws + (size_t)BSZ * SL * DIM * sizeof(float));

  const dim3 blk(256);
  ar_kernel<<<dim3(SL / 128, DIM / 64, BSZ), blk, 0, stream>>>(u, mu, deltas);
  for (int chunk = 0; chunk < NCHUNK; ++chunk) {
    conv_kernel<<<dim3(SL / 128, DIM / 64, BSZ * NFC), blk, 0, stream>>>(
        u, evec, eval, xt, chunk);
    gemm_phi_kernel<<<dim3(BSZ * SL / 128, DIM / 64, 1), blk, 0, stream>>>(
        xt, mphi, deltas, chunk);
  }
  scan_kernel<<<dim3(BSZ), blk, 0, stream>>>(deltas, my, out);
}
